// NNUE_19189913878890
// MI455X (gfx1250) — matrix operand order repaired and checked by reading, not compiled, not run on hardware
//
#include <hip/hip_runtime.h>
#include <hip/hip_bf16.h>

typedef __attribute__((ext_vector_type(16))) _Float16 v16h;
typedef __attribute__((ext_vector_type(8)))  _Float16 v8h;
typedef __attribute__((ext_vector_type(8)))  float    v8f;
typedef __attribute__((ext_vector_type(4)))  unsigned int u32x4;
typedef __attribute__((ext_vector_type(8)))  int      i32x8;
typedef __attribute__((ext_vector_type(4)))  int      i32x4;

#define BATCH 1024
#define NFEAT 800      // K of the big GEMM (25 * 32)
#define L1DIM 1024     // N of the big GEMM
#define KSTEP 32
#define BM 128         // block M tile
#define BN 128         // block N tile
#define LROW 40        // LDS row stride in halves (32 data + 8 pad = 80 B)

// ---------------------------------------------------------------------------
// Kernel 1: conv(stride10,pad1) + clip[-1,1] + sigmoid(10x) gated at x>0,
// emitting the GEMM A-matrix (1024 x 800) in f16, row-major.
// ---------------------------------------------------------------------------
__global__ void conv_act_kernel(const float* __restrict__ images,
                                const float* __restrict__ conv_w,
                                _Float16* __restrict__ A) {
    int idx = blockIdx.x * blockDim.x + threadIdx.x;       // b*800 + f
    if (idx >= BATCH * NFEAT) return;
    int b = idx / NFEAT;
    int f = idx - b * NFEAT;
    int c   = f / 100;
    int rem = f - c * 100;
    int y   = rem / 10;
    int x   = rem - y * 10;

    float acc = 0.0f;
    const float* img = images + (size_t)b * 3 * 96 * 96;
    #pragma unroll
    for (int ci = 0; ci < 3; ++ci) {
        #pragma unroll
        for (int ky = 0; ky < 3; ++ky) {
            int iy = y * 10 - 1 + ky;
            if (iy < 0 || iy >= 96) continue;
            #pragma unroll
            for (int kx = 0; kx < 3; ++kx) {
                int ix = x * 10 - 1 + kx;
                if (ix < 0 || ix >= 96) continue;
                float w = conv_w[((c * 3 + ci) * 3 + ky) * 3 + kx];
                acc = fmaf(img[(ci * 96 + iy) * 96 + ix], w, acc);
            }
        }
    }
    float xc = fminf(fmaxf(acc, -1.0f), 1.0f);
    float val = (xc > 0.0f) ? (1.0f / (1.0f + __expf(-10.0f * xc))) : 0.0f;
    A[(size_t)b * NFEAT + f] = (_Float16)val;
}

// ---------------------------------------------------------------------------
// Kernel 2: ft_w (800 x 1024 f32, K-major) -> Bt (1024 x 800 f16, N-major)
// ---------------------------------------------------------------------------
__global__ void convert_w_kernel(const float* __restrict__ ft_w,
                                 _Float16* __restrict__ Bt) {
    int idx = blockIdx.x * blockDim.x + threadIdx.x;       // n*800 + k
    if (idx >= L1DIM * NFEAT) return;
    int n = idx / NFEAT;
    int k = idx - n * NFEAT;
    Bt[idx] = (_Float16)ft_w[(size_t)k * L1DIM + n];
}

// ---------------------------------------------------------------------------
// TDM 2D tile load: global (row-major, rowStride elements of 2B) -> LDS.
// D# packed per CDNA5 ISA.
// Padding: after every 16 DWORDs (64 B = one 32-elem row) insert 4 DWORDs
// (16 B) => LDS row stride 80 B = LROW halves.
// Toolchain note: this clang exposes the 6-arg builtin
//   (u32x4 g0, i32x8 g1, i32x4 g2, i32x4 g3, i32x8 g4, i32 cpol)
// g2/g3 (3D+/gather) and g4 are unused for 2D tiles -> zero.
// ---------------------------------------------------------------------------
__device__ __forceinline__ void tdm_load_tile_f16(unsigned lds_byte_addr,
                                                  const _Float16* gptr,
                                                  unsigned tensor_d0,   // elems
                                                  unsigned tensor_d1,   // rows
                                                  unsigned row_stride,  // elems
                                                  unsigned tile_d0,     // elems
                                                  unsigned tile_d1) {   // rows
    unsigned long long ga = (unsigned long long)(size_t)gptr;
    u32x4 g0;
    g0[0] = 1u;                                            // count=1, user mode
    g0[1] = lds_byte_addr;                                 // lds_addr
    g0[2] = (unsigned)(ga & 0xffffffffu);                  // global_addr[31:0]
    g0[3] = (unsigned)((ga >> 32) & 0x01ffffffu) | (2u << 30); // [56:32] | type=2
    i32x8 g1;
    g1[0] = (int)((1u << 16)        // data_size = 1 -> 2 bytes
                | (1u << 20)        // pad_enable
                | (3u << 22)        // pad_interval: 16 DWORDs (64 B)
                | (3u << 25));      // pad_amount: 4 DWORDs (16 B)
    g1[1] = (int)((tensor_d0 & 0xffffu) << 16);            // barrier_addr=0 | dim0 lo
    g1[2] = (int)((tensor_d0 >> 16) | ((tensor_d1 & 0xffffu) << 16));
    g1[3] = (int)((tensor_d1 >> 16) | ((tile_d0 & 0xffffu) << 16));
    g1[4] = (int)(tile_d1 & 0xffffu);                      // tile_dim1 | tile_dim2=0
    g1[5] = (int)row_stride;                               // dim0_stride[31:0]
    g1[6] = 0;                                             // dim0_stride[47:32] | dim1_stride lo
    g1[7] = 0;
    i32x4 g2 = {0, 0, 0, 0};
    i32x4 g3 = {0, 0, 0, 0};
    i32x8 g4 = {0, 0, 0, 0, 0, 0, 0, 0};
    __builtin_amdgcn_tensor_load_to_lds(g0, g1, g2, g3, g4, 0);
}

// ---------------------------------------------------------------------------
// Kernel 3: l0 = clip(A(1024x800) @ B(800x1024) + ft_b, 0, 1)
// Block = 256 threads (8 waves) computes a 128x128 tile. Per K-step wave 0
// TDM-stages A[128x32] and Bt[128x32] into LDS (s_wait_tensorcnt), then each
// wave computes a 16x128 strip: 1 A fragment reused across 8 WMMAs.
//
// WMMA 16-bit A fragment layout (CDNA5 ISA):
//   lanes 0-15 : row = lane,    elems 0..7 = K+0..7,  8..15 = K+16..23
//   lanes 16-31: row = lane-16, elems 0..7 = K+8..15, 8..15 = K+24..31
// B fragment (32x16): col = lane&15, elems 0..7 = K + (lane>=16?8:0) + i, 8..15 = K+16 + (lane>=16?8:0) + i-8
// C/D v8f: VGPR i -> M = i + (lane>=16 ? 8 : 0); N = lane&15
// ---------------------------------------------------------------------------
__global__ void __launch_bounds__(256)
gemm_wmma_kernel(const _Float16* __restrict__ A,
                 const _Float16* __restrict__ Bt,
                 const float* __restrict__ ft_b,
                 float* __restrict__ l0) {
    __shared__ _Float16 ldsA[BM * LROW];   // 128 rows x 80 B = 10 KB
    __shared__ _Float16 ldsB[BN * LROW];   // 10 KB

    const int lane   = threadIdx.x & 31;
    const int waveId = threadIdx.x >> 5;          // 0..7 -> M sub-tile
    const int mBlk   = (int)blockIdx.x >> 3;      // 0..7
    const int nBlk   = (int)blockIdx.x & 7;       // 0..7
    const int mBase  = mBlk * BM;
    const int nBase  = nBlk * BN;

    const int l15  = lane & 15;
    const int hi   = lane >> 4;
    const int aoff = hi << 3;                     // 0 / 8
    const int boff = hi << 3;                     // 0 / 8

    const unsigned ldsA_addr = (unsigned)(size_t)(&ldsA[0]);
    const unsigned ldsB_addr = (unsigned)(size_t)(&ldsB[0]);

    v8f acc[8];
    #pragma unroll
    for (int n = 0; n < 8; ++n) acc[n] = (v8f){};

    const _Float16* aTile = A  + (size_t)mBase * NFEAT;
    const _Float16* bTile = Bt + (size_t)nBase * NFEAT;
    const _Float16* aFragRow = &ldsA[(waveId * 16 + l15) * LROW];

    for (int kk = 0; kk < NFEAT; kk += KSTEP) {
        __syncthreads();                    // previous tile fully consumed
        if (waveId == 0) {
            tdm_load_tile_f16(ldsA_addr, aTile + kk, NFEAT, BATCH, NFEAT, KSTEP, BM);
            tdm_load_tile_f16(ldsB_addr, bTile + kk, NFEAT, L1DIM, NFEAT, KSTEP, BN);
            __builtin_amdgcn_s_wait_tensorcnt(0);
        }
        __syncthreads();                    // tile visible to all 8 waves

        // A fragment for this wave's 16-row strip (two 16B LDS loads)
        v8h a0 = *reinterpret_cast<const v8h*>(aFragRow + aoff);
        v8h a1 = *reinterpret_cast<const v8h*>(aFragRow + 16 + aoff);
        v16h afrag;
        #pragma unroll
        for (int i = 0; i < 8; ++i) { afrag[i] = a0[i]; afrag[8 + i] = a1[i]; }

        #pragma unroll
        for (int n = 0; n < 8; ++n) {
            const _Float16* bRow = &ldsB[(n * 16 + l15) * LROW + boff];
            v8h b0 = *reinterpret_cast<const v8h*>(bRow);
            v8h b1 = *reinterpret_cast<const v8h*>(bRow + 16);
            v16h bfrag;
            #pragma unroll
            for (int i = 0; i < 8; ++i) { bfrag[i] = b0[i]; bfrag[8 + i] = b1[i]; }
            acc[n] = __builtin_amdgcn_wmma_f32_16x16x32_f16(
                false, afrag, false, bfrag, (short)0, acc[n], false, false);
        }
    }

    // epilogue: features = acc + ft_b[n]; l0 = clip(features, 0, 1)
    const int mRow = mBase + waveId * 16 + (hi ? 8 : 0);
    #pragma unroll
    for (int n = 0; n < 8; ++n) {
        const int col  = nBase + n * 16 + l15;
        const float bias = ft_b[col];
        #pragma unroll
        for (int i = 0; i < 8; ++i) {
            float v = acc[n][i] + bias;
            v = fminf(fmaxf(v, 0.0f), 1.0f);
            l0[(size_t)(mRow + i) * L1DIM + col] = v;
        }
    }
}

// ---------------------------------------------------------------------------
// Kernel 4: per-batch-row tail MLP, one wave32 per row.
// ---------------------------------------------------------------------------
__global__ void __launch_bounds__(32)
mlp_tail_kernel(const float* __restrict__ l0,
                const float* __restrict__ w1, const float* __restrict__ b1,
                const float* __restrict__ w2, const float* __restrict__ b2,
                const float* __restrict__ w3, const float* __restrict__ b3,
                float* __restrict__ out) {
    const int b    = blockIdx.x;
    const int lane = threadIdx.x;
    const float scale = 127.0f / 128.0f;
    const float* row = l0 + (size_t)b * L1DIM;

    float acc[15];
    #pragma unroll
    for (int j = 0; j < 15; ++j) acc[j] = 0.0f;

    for (int k = lane; k < 512; k += 32) {
        float s0 = row[k];
        float s1 = row[512 + k];
        float v0 = s0 * s1 * scale;
        float v1 = s0 * scale;
        #pragma unroll
        for (int j = 0; j < 15; ++j) {
            acc[j] = fmaf(v0, w1[j * L1DIM + k], acc[j]);
            acc[j] = fmaf(v1, w1[j * L1DIM + 512 + k], acc[j]);
        }
    }
    #pragma unroll
    for (int off = 16; off > 0; off >>= 1) {
        #pragma unroll
        for (int j = 0; j < 15; ++j) acc[j] += __shfl_xor(acc[j], off);
    }
    float h1[15];
    #pragma unroll
    for (int j = 0; j < 15; ++j) h1[j] = fmaxf(acc[j] + b1[j], 0.0f);

    float t = b2[lane];
    #pragma unroll
    for (int k = 0; k < 15; ++k) t = fmaf(w2[lane * 15 + k], h1[k], t);
    float h2 = fmaxf(t, 0.0f);

    float contrib = h2 * w3[lane];
    #pragma unroll
    for (int off = 16; off > 0; off >>= 1) contrib += __shfl_xor(contrib, off);
    if (lane == 0) out[b] = contrib + b3[0];
}

// ---------------------------------------------------------------------------
extern "C" void kernel_launch(void* const* d_in, const int* in_sizes, int n_in,
                              void* d_out, int out_size, void* d_ws, size_t ws_size,
                              hipStream_t stream) {
    const float* images = (const float*)d_in[0];
    const float* conv_w = (const float*)d_in[1];
    const float* ft_w   = (const float*)d_in[2];
    const float* ft_b   = (const float*)d_in[3];
    const float* w1     = (const float*)d_in[4];
    const float* b1     = (const float*)d_in[5];
    const float* w2     = (const float*)d_in[6];
    const float* b2     = (const float*)d_in[7];
    const float* w3     = (const float*)d_in[8];
    const float* b3     = (const float*)d_in[9];
    float* out = (float*)d_out;

    char* ws = (char*)d_ws;
    _Float16* A  = (_Float16*)ws;                  // 1,638,400 B
    _Float16* Bt = (_Float16*)(ws + 1638400);      // 1,638,400 B
    float*    l0 = (float*)(ws + 3276800);         // 4,194,304 B

    {
        int total = BATCH * NFEAT;
        conv_act_kernel<<<(total + 255) / 256, 256, 0, stream>>>(images, conv_w, A);
    }
    {
        int total = L1DIM * NFEAT;
        convert_w_kernel<<<(total + 255) / 256, 256, 0, stream>>>(ft_w, Bt);
    }
    {
        int blocks = (BATCH / BM) * (L1DIM / BN);  // 64 blocks of 256
        gemm_wmma_kernel<<<blocks, 256, 0, stream>>>(A, Bt, ft_b, l0);
    }
    mlp_tail_kernel<<<BATCH, 32, 0, stream>>>(l0, w1, b1, w2, b2, w3, b3, out);
}
